// Transformer_27642409517555
// MI455X (gfx1250) — matrix operand order repaired and checked by reading, not compiled, not run on hardware
//
#include <hip/hip_runtime.h>
#include <stdint.h>

// ---------------- model constants ----------------
#define DIMX    1024
#define NHEADS  16
#define HDIM    64
#define HIDDEN  2816
#define SEQ     2048
#define BATCH   4
#define TOKENS  (BATCH * SEQ)
#define WINDOW  128

// ---------------- CDNA5 async global->LDS (gated; fallback = sync path) ----
#if defined(__has_builtin)
# if __has_builtin(__builtin_amdgcn_global_load_async_to_lds_b128)
#  define HAVE_ASYNC_LDS 1
# else
#  define HAVE_ASYNC_LDS 0
# endif
# if __has_builtin(__builtin_amdgcn_s_wait_asynccnt)
#  define HAVE_WAIT_ASYNC 1
# else
#  define HAVE_WAIT_ASYNC 0
# endif
#else
# define HAVE_ASYNC_LDS 0
# define HAVE_WAIT_ASYNC 0
#endif

typedef __attribute__((ext_vector_type(4))) int i32x4;

#if HAVE_ASYNC_LDS
// param0: <4 x i32> addrspace(1)* ("int4 __device__*" per clang diagnostic)
// param1: <4 x i32> addrspace(3)*
typedef __attribute__((address_space(1))) i32x4 gi32x4;
typedef __attribute__((address_space(3))) i32x4 li32x4;
__device__ __forceinline__ void async_b128(const void* g, void* l) {
  __builtin_amdgcn_global_load_async_to_lds_b128((gi32x4*)g, (li32x4*)l, 0, 0);
}
#endif

__device__ __forceinline__ void wait_async0() {
#if HAVE_ASYNC_LDS
# if HAVE_WAIT_ASYNC
  __builtin_amdgcn_s_wait_asynccnt(0);
# else
  asm volatile("s_wait_asynccnt 0" ::: "memory");
# endif
#endif
}

// ---------------- vector / fragment types ----------------
typedef __attribute__((ext_vector_type(16))) __bf16    v16bf;
typedef __attribute__((ext_vector_type(16))) _Float16  v16h;
typedef __attribute__((ext_vector_type(8)))  float     v8f;
typedef __attribute__((ext_vector_type(4)))  unsigned  u32x4;
typedef __attribute__((ext_vector_type(2)))  unsigned  u32x2;
typedef __attribute__((ext_vector_type(4)))  float     f32x4;

union FragBF { u32x4 q[2]; v16bf v; };   // 16 bf16 = 8 VGPRs
union FragH  { u32x4 q[2]; v16h  v; };   // 16 f16  = 8 VGPRs

__device__ __forceinline__ unsigned short f2bf(float f) {
  union { float f; unsigned u; } v; v.f = f;
  unsigned r = v.u + 0x7fffu + ((v.u >> 16) & 1u);   // RNE
  return (unsigned short)(r >> 16);
}

// ---------------- fp32 -> bf16 weight convert ----------------
__global__ void cvt_f32_bf16(const float* __restrict__ src,
                             unsigned short* __restrict__ dst, int n) {
  int i = blockIdx.x * blockDim.x + threadIdx.x;
  if (i < n) dst[i] = f2bf(src[i]);
}

// ---------------- LayerNorm (fp32 in, bf16 out) ----------------
__global__ __launch_bounds__(256) void layernorm_bf16(
    const float* __restrict__ X, const float* __restrict__ G,
    const float* __restrict__ Bb, unsigned short* __restrict__ O) {
  int row = blockIdx.x;
  int t = threadIdx.x;
  const float* xr = X + (size_t)row * DIMX;
  f32x4 xv = *(const f32x4*)(xr + t * 4);
  float s  = xv[0] + xv[1] + xv[2] + xv[3];
  float s2 = xv[0]*xv[0] + xv[1]*xv[1] + xv[2]*xv[2] + xv[3]*xv[3];
  __shared__ float rs[8], rs2[8];
#pragma unroll
  for (int m = 16; m >= 1; m >>= 1) { s += __shfl_xor(s, m, 32); s2 += __shfl_xor(s2, m, 32); }
  if ((t & 31) == 0) { rs[t >> 5] = s; rs2[t >> 5] = s2; }
  __syncthreads();
  float ts = 0.f, ts2 = 0.f;
#pragma unroll
  for (int i = 0; i < 8; i++) { ts += rs[i]; ts2 += rs2[i]; }
  float mu  = ts * (1.0f / DIMX);
  float var = ts2 * (1.0f / DIMX) - mu * mu;
  float inv = rsqrtf(var + 1e-5f);
#pragma unroll
  for (int i = 0; i < 4; i++) {
    int c = t * 4 + i;
    O[(size_t)row * DIMX + c] = f2bf((xv[i] - mu) * inv * G[c] + Bb[c]);
  }
}

// ---------------- generic bf16 WMMA GEMM ----------------
// C[M,N] = A[M,K] @ B[K,N]; A,B bf16 row-major.
// EPI: 0=store f32, 1=store f16, 2=store bf16, 3=f32 + residual
// DUAL: two B matrices, epilogue silu(acc0)*acc1 -> bf16 (SwiGLU up-proj)
template <int EPI, bool DUAL>
__global__ __launch_bounds__(256) void gemm_bf16_wmma(
    const unsigned short* __restrict__ A, const unsigned short* __restrict__ B0,
    const unsigned short* __restrict__ B1, void* __restrict__ Cout,
    const float* __restrict__ resid, int M, int N, int K) {
  constexpr int BM = 128, BN = 64, BK = 32, LDA = 56, LDB = 56; // 112B strides, 16B aligned
  __shared__ alignas(16) unsigned short As[BM][LDA];
  __shared__ alignas(16) unsigned short Bs0[BN][LDB];                 // transposed: Bs0[n][k]
  __shared__ alignas(16) unsigned short Bs1[DUAL ? BN : 1][LDB];

  const int t = threadIdx.x;
  const int wave = t >> 5, lane = t & 31;
  const int g = lane >> 4, ln = lane & 15;
  const int wm = (wave & 3) * 32, wn = (wave >> 2) * 32;
  const int rowBase = blockIdx.y * BM, colBase = blockIdx.x * BN;

  v8f acc0[2][2] = {};
  v8f acc1[2][2] = {};

  for (int k0 = 0; k0 < K; k0 += BK) {
    __syncthreads();
    // ---- stage A tile (row-major): async DMA to LDS when available ----
#if HAVE_ASYNC_LDS
#pragma unroll
    for (int i = 0; i < 2; i++) {
      int cid = t + i * 256;                  // 512 chunks of 8 bf16 (16B)
      int r = cid >> 2, c8 = (cid & 3) * 8;
      async_b128(&A[(size_t)(rowBase + r) * K + k0 + c8], &As[r][c8]);
    }
#else
#pragma unroll
    for (int i = 0; i < 4; i++) {
      int cid = t + i * 256;                  // 1024 chunks of 4 bf16
      int r = cid >> 3, c4 = (cid & 7) * 4;
      *(u32x2*)&As[r][c4] =
          *(const u32x2*)&A[(size_t)(rowBase + r) * K + k0 + c4];
    }
#endif
    // ---- stage B tiles transposed (Bs[n][k]) so fragments are contiguous ----
#pragma unroll
    for (int i = 0; i < 4; i++) {
      int cid = t + i * 256;                  // 1024 chunks of 2 bf16 along N
      int kk = cid >> 5, n2 = (cid & 31) * 2;
      unsigned v0 = *(const unsigned*)&B0[(size_t)(k0 + kk) * N + colBase + n2];
      Bs0[n2][kk]     = (unsigned short)(v0 & 0xffffu);
      Bs0[n2 + 1][kk] = (unsigned short)(v0 >> 16);
      if constexpr (DUAL) {
        unsigned v1 = *(const unsigned*)&B1[(size_t)(k0 + kk) * N + colBase + n2];
        Bs1[n2][kk]     = (unsigned short)(v1 & 0xffffu);
        Bs1[n2 + 1][kk] = (unsigned short)(v1 >> 16);
      }
    }
    // ---- prefetch next K-tile into cache while WMMAs run ----
    if (k0 + BK < K) {
      int r = t >> 1, c16 = (t & 1) * 16;
      __builtin_prefetch(&A[(size_t)(rowBase + r) * K + k0 + BK + c16], 0, 1);
      int kk = t >> 3, n8 = (t & 7) * 8;
      __builtin_prefetch(&B0[(size_t)(k0 + BK + kk) * N + colBase + n8], 0, 1);
      if constexpr (DUAL)
        __builtin_prefetch(&B1[(size_t)(k0 + BK + kk) * N + colBase + n8], 0, 1);
    }
    wait_async0();
    __syncthreads();

    // ---- fragments (CDNA5 wave32 WMMA layouts) ----
    FragBF af[2], bf0[2], bf1[2];
#pragma unroll
    for (int mi = 0; mi < 2; mi++) {           // A: m=ln, k = g*16 + ...
      const unsigned short* p = &As[wm + mi * 16 + ln][g * 16];
      af[mi].q[0] = *(const u32x4*)p;
      af[mi].q[1] = *(const u32x4*)(p + 8);
    }
#pragma unroll
    for (int ni = 0; ni < 2; ni++) {           // B: n=ln, k = g*16 + 2v (+1)
      const unsigned short* p = &Bs0[wn + ni * 16 + ln][g * 16];
      bf0[ni].q[0] = *(const u32x4*)p;
      bf0[ni].q[1] = *(const u32x4*)(p + 8);
      if constexpr (DUAL) {
        const unsigned short* p1 = &Bs1[wn + ni * 16 + ln][g * 16];
        bf1[ni].q[0] = *(const u32x4*)p1;
        bf1[ni].q[1] = *(const u32x4*)(p1 + 8);
      }
    }
#pragma unroll
    for (int mi = 0; mi < 2; mi++)
#pragma unroll
      for (int ni = 0; ni < 2; ni++) {
        acc0[mi][ni] = __builtin_amdgcn_wmma_f32_16x16x32_bf16(
            false, af[mi].v, false, bf0[ni].v, (short)0, acc0[mi][ni], false, false);
        if constexpr (DUAL)
          acc1[mi][ni] = __builtin_amdgcn_wmma_f32_16x16x32_bf16(
              false, af[mi].v, false, bf1[ni].v, (short)0, acc1[mi][ni], false, false);
      }
  }

  // ---- epilogue (C layout: row = g*8+v, col = ln) ----
#pragma unroll
  for (int mi = 0; mi < 2; mi++)
#pragma unroll
    for (int ni = 0; ni < 2; ni++)
#pragma unroll
      for (int v = 0; v < 8; v++) {
        int row = rowBase + wm + mi * 16 + g * 8 + v;
        int col = colBase + wn + ni * 16 + ln;
        size_t idx = (size_t)row * N + col;
        float a = acc0[mi][ni][v];
        if constexpr (DUAL) {
          float b = acc1[mi][ni][v];
          float si = a / (1.0f + __expf(-a));           // silu
          ((unsigned short*)Cout)[idx] = f2bf(si * b);
        } else if constexpr (EPI == 0) {
          ((float*)Cout)[idx] = a;
        } else if constexpr (EPI == 1) {
          ((_Float16*)Cout)[idx] = (_Float16)a;
        } else if constexpr (EPI == 2) {
          ((unsigned short*)Cout)[idx] = f2bf(a);
        } else {
          ((float*)Cout)[idx] = a + resid[idx];
        }
      }
}

// ---------------- RoPE (in-place on f16 Q and K) ----------------
__global__ __launch_bounds__(256) void rope_f16(_Float16* __restrict__ Q,
                                                _Float16* __restrict__ Kp) {
  int id = blockIdx.x * blockDim.x + threadIdx.x;   // TOKENS*NHEADS*32
  int pair = id & 31;
  int h = (id >> 5) & (NHEADS - 1);
  int tok = id >> 9;
  int s = tok & (SEQ - 1);
  float freq = __expf(-(float)pair * (1.0f / 32.0f) * 9.210340371976184f); // ln(1e4)
  float ang = (float)s * freq;
  float sn, cs;
  __sincosf(ang, &sn, &cs);
  size_t base = ((size_t)tok * NHEADS + h) * HDIM + pair * 2;
  float qr = (float)Q[base], qi = (float)Q[base + 1];
  Q[base]     = (_Float16)(qr * cs - qi * sn);
  Q[base + 1] = (_Float16)(qr * sn + qi * cs);
  float kr = (float)Kp[base], ki = (float)Kp[base + 1];
  Kp[base]     = (_Float16)(kr * cs - ki * sn);
  Kp[base + 1] = (_Float16)(kr * sn + ki * cs);
}

// ---------------- flash-style sliding-window attention ----------------
// block = (128 queries) x (1 head) x (1 batch); 8 waves, 16 query rows each.
// keys of a 128-query block live in exactly two 128-blocks -> 8 chunks of 32.
__global__ __launch_bounds__(256) void attention_wmma(
    const _Float16* __restrict__ Q, const _Float16* __restrict__ Kg,
    const _Float16* __restrict__ Vg, unsigned short* __restrict__ Og) {
  constexpr int LQ = 72, LK = 72, LV = 48, LP = 48;   // 16B-aligned row strides
  __shared__ alignas(16) _Float16 Qs[128][LQ];
  __shared__ alignas(16) _Float16 Ks[32][LK];
  __shared__ alignas(16) _Float16 Vt[64][LV];         // transposed: Vt[d][key]
  __shared__ alignas(16) _Float16 Ps[8][16][LP];      // per-wave P tile

  const int qb = blockIdx.x, h = blockIdx.y, b = blockIdx.z;
  const int t = threadIdx.x, wave = t >> 5, lane = t & 31;
  const int g = lane >> 4, ln = lane & 15;

  // stage Q block (128 x 64) -- async when available; waited before 1st consume
#if HAVE_ASYNC_LDS
#pragma unroll
  for (int i = 0; i < 4; i++) {
    int cid = t + i * 256;                   // 1024 chunks of 8 f16 (16B)
    int r = cid >> 3, c8 = (cid & 7) * 8;
    async_b128(&Q[(((size_t)(b * SEQ + qb * 128 + r)) * NHEADS + h) * HDIM + c8],
               &Qs[r][c8]);
  }
#else
#pragma unroll
  for (int i = 0; i < 8; i++) {
    int cid = t + i * 256;
    int r = cid >> 4, c4 = (cid & 15) * 4;
    *(u32x2*)&Qs[r][c4] = *(const u32x2*)
        &Q[(((size_t)(b * SEQ + qb * 128 + r)) * NHEADS + h) * HDIM + c4];
  }
#endif

  float mrow[8], lrow[8];
  v8f oacc[4] = {};
#pragma unroll
  for (int v = 0; v < 8; v++) { mrow[v] = -3.0e38f; lrow[v] = 0.f; }

  const int iMin = qb * 128 + wave * 16;
  const int c0 = (qb == 0) ? 4 : 0;          // block-uniform skip of negative keys
  for (int c = c0; c < 8; c++) {
    int ks0 = qb * 128 - 128 + c * 32;
    __syncthreads();                         // prev-iter PV reads complete
    // stage K chunk (row-major, async) + V chunk (transposed, scalar)
#if HAVE_ASYNC_LDS
    {
      int r = t >> 3, c8 = (t & 7) * 8;      // 256 chunks of 8 f16 (16B)
      async_b128(&Kg[(((size_t)(b * SEQ + ks0 + r)) * NHEADS + h) * HDIM + c8],
                 &Ks[r][c8]);
    }
#else
#pragma unroll
    for (int i = 0; i < 2; i++) {
      int cid = t + i * 256;
      int r = cid >> 4, c4 = (cid & 15) * 4;
      *(u32x2*)&Ks[r][c4] = *(const u32x2*)
          &Kg[(((size_t)(b * SEQ + ks0 + r)) * NHEADS + h) * HDIM + c4];
    }
#endif
#pragma unroll
    for (int i = 0; i < 8; i++) {
      int e = t + i * 256;
      int key = e >> 6, d = e & 63;
      Vt[d][key] = Vg[(((size_t)(b * SEQ + ks0 + key)) * NHEADS + h) * HDIM + d];
    }
    wait_async0();                           // covers Q tile + this K tile
    __syncthreads();

    // wave-uniform skip of fully-masked chunks (saves ~40% of WMMAs)
    bool valid = (ks0 <= iMin + 15) && (ks0 + 31 >= iMin - WINDOW);
    float p0[8], p1[8];
    if (valid) {
      // S = Q K^T : two 16x16 key sub-tiles, k-loop over d=64 in two steps
      v8f sc[2] = {};
      FragH aq[2];
#pragma unroll
      for (int kb = 0; kb < 2; kb++) {
        const _Float16* p = &Qs[wave * 16 + ln][kb * 32 + g * 16];
        aq[kb].q[0] = *(const u32x4*)p;
        aq[kb].q[1] = *(const u32x4*)(p + 8);
      }
#pragma unroll
      for (int sub = 0; sub < 2; sub++)
#pragma unroll
        for (int kb = 0; kb < 2; kb++) {
          FragH bk;
          const _Float16* p = &Ks[sub * 16 + ln][kb * 32 + g * 16];
          bk.q[0] = *(const u32x4*)p;
          bk.q[1] = *(const u32x4*)(p + 8);
          sc[sub] = __builtin_amdgcn_wmma_f32_16x16x32_f16(
              false, aq[kb].v, false, bk.v, (short)0, sc[sub], false, false);
        }

      // online softmax (f32), rows = g*8+v, cols = ln within half
#pragma unroll
      for (int v = 0; v < 8; v++) {
        int i = iMin + g * 8 + v;
        int j0 = ks0 + ln, j1 = ks0 + 16 + ln;
        float s0 = sc[0][v] * 0.125f;
        float s1 = sc[1][v] * 0.125f;
        if (!(j0 <= i && i - j0 <= WINDOW)) s0 = -3.0e38f;
        if (!(j1 <= i && i - j1 <= WINDOW)) s1 = -3.0e38f;
        float cm = fmaxf(s0, s1);
#pragma unroll
        for (int m = 8; m >= 1; m >>= 1) cm = fmaxf(cm, __shfl_xor(cm, m, 32));
        float mnew = fmaxf(mrow[v], cm);
        float mn = fmaxf(mnew, -1.0e30f);     // guard fully-masked rows
        float fac = __expf(mrow[v] - mn);
        float e0 = __expf(s0 - mn), e1 = __expf(s1 - mn);
        float rs = e0 + e1;
#pragma unroll
        for (int m = 8; m >= 1; m >>= 1) rs += __shfl_xor(rs, m, 32);
        lrow[v] = lrow[v] * fac + rs;
        mrow[v] = mnew;
#pragma unroll
        for (int n4 = 0; n4 < 4; n4++) oacc[n4][v] *= fac;
        p0[v] = e0; p1[v] = e1;
      }
      // stage P (C-layout -> row-major LDS, reread in A-layout)
#pragma unroll
      for (int v = 0; v < 8; v++) {
        Ps[wave][g * 8 + v][ln]      = (_Float16)p0[v];
        Ps[wave][g * 8 + v][16 + ln] = (_Float16)p1[v];
      }
    }
    __syncthreads();
    if (valid) {
      FragH ap;
      const _Float16* pp = &Ps[wave][ln][g * 16];
      ap.q[0] = *(const u32x4*)pp;
      ap.q[1] = *(const u32x4*)(pp + 8);
#pragma unroll
      for (int n4 = 0; n4 < 4; n4++) {        // O += P @ V over d sub-tiles
        FragH bv;
        const _Float16* pv = &Vt[n4 * 16 + ln][g * 16];
        bv.q[0] = *(const u32x4*)pv;
        bv.q[1] = *(const u32x4*)(pv + 8);
        oacc[n4] = __builtin_amdgcn_wmma_f32_16x16x32_f16(
            false, ap.v, false, bv.v, (short)0, oacc[n4], false, false);
      }
    }
  }

  // write O / l  -> bf16 [B,S,DIM] (heads concatenated)
#pragma unroll
  for (int n4 = 0; n4 < 4; n4++)
#pragma unroll
    for (int v = 0; v < 8; v++) {
      int i = iMin + g * 8 + v;
      int d = n4 * 16 + ln;
      float o = oacc[n4][v] / lrow[v];
      Og[((size_t)(b * SEQ + i)) * DIMX + h * HDIM + d] = f2bf(o);
    }
}

// ---------------- host orchestration ----------------
extern "C" void kernel_launch(void* const* d_in, const int* in_sizes, int n_in,
                              void* d_out, int out_size, void* d_ws, size_t ws_size,
                              hipStream_t stream) {
  const float* x     = (const float*)d_in[0];
  const float* wq    = (const float*)d_in[1];
  const float* wk    = (const float*)d_in[2];
  const float* wv    = (const float*)d_in[3];
  const float* wo    = (const float*)d_in[4];
  const float* w1    = (const float*)d_in[5];
  const float* w2    = (const float*)d_in[6];
  const float* w3    = (const float*)d_in[7];
  const float* ln1g  = (const float*)d_in[8];
  const float* ln1b  = (const float*)d_in[9];
  const float* ln2g  = (const float*)d_in[10];
  const float* ln2b  = (const float*)d_in[11];

  unsigned char* ws = (unsigned char*)d_ws;
  size_t off = 0;
  auto take = [&](size_t bytes) -> void* {
    void* p = ws + off;
    off = (off + bytes + 255) & ~(size_t)255;
    return p;
  };
  unsigned short* wqb = (unsigned short*)take((size_t)DIMX * DIMX * 2);
  unsigned short* wkb = (unsigned short*)take((size_t)DIMX * DIMX * 2);
  unsigned short* wvb = (unsigned short*)take((size_t)DIMX * DIMX * 2);
  unsigned short* wob = (unsigned short*)take((size_t)DIMX * DIMX * 2);
  unsigned short* w1b = (unsigned short*)take((size_t)DIMX * HIDDEN * 2);
  unsigned short* w3b = (unsigned short*)take((size_t)DIMX * HIDDEN * 2);
  unsigned short* w2b = (unsigned short*)take((size_t)HIDDEN * DIMX * 2);
  unsigned short* xn  = (unsigned short*)take((size_t)TOKENS * DIMX * 2);
  _Float16*       qf  = (_Float16*)take((size_t)TOKENS * DIMX * 2);
  _Float16*       kf  = (_Float16*)take((size_t)TOKENS * DIMX * 2);
  _Float16*       vf  = (_Float16*)take((size_t)TOKENS * DIMX * 2);
  unsigned short* ao  = (unsigned short*)take((size_t)TOKENS * DIMX * 2);
  float*          hbuf= (float*)take((size_t)TOKENS * DIMX * 4);
  unsigned short* hn  = (unsigned short*)take((size_t)TOKENS * DIMX * 2);
  unsigned short* gact= (unsigned short*)take((size_t)TOKENS * HIDDEN * 2);
  (void)ws_size; (void)n_in; (void)in_sizes; (void)out_size;

  // weights -> bf16 (stay L2-resident thereafter)
  int n1 = DIMX * DIMX, n2 = DIMX * HIDDEN;
  cvt_f32_bf16<<<(n1 + 255) / 256, 256, 0, stream>>>(wq, wqb, n1);
  cvt_f32_bf16<<<(n1 + 255) / 256, 256, 0, stream>>>(wk, wkb, n1);
  cvt_f32_bf16<<<(n1 + 255) / 256, 256, 0, stream>>>(wv, wvb, n1);
  cvt_f32_bf16<<<(n1 + 255) / 256, 256, 0, stream>>>(wo, wob, n1);
  cvt_f32_bf16<<<(n2 + 255) / 256, 256, 0, stream>>>(w1, w1b, n2);
  cvt_f32_bf16<<<(n2 + 255) / 256, 256, 0, stream>>>(w3, w3b, n2);
  cvt_f32_bf16<<<(n2 + 255) / 256, 256, 0, stream>>>(w2, w2b, n2);

  // LN1
  layernorm_bf16<<<TOKENS, 256, 0, stream>>>(x, ln1g, ln1b, xn);

  // Q/K/V projections (bf16 WMMA, f16 out for attention)
  dim3 gproj(DIMX / 64, TOKENS / 128);
  gemm_bf16_wmma<1, false><<<gproj, 256, 0, stream>>>(xn, wqb, nullptr, qf, nullptr, TOKENS, DIMX, DIMX);
  gemm_bf16_wmma<1, false><<<gproj, 256, 0, stream>>>(xn, wkb, nullptr, kf, nullptr, TOKENS, DIMX, DIMX);
  gemm_bf16_wmma<1, false><<<gproj, 256, 0, stream>>>(xn, wvb, nullptr, vf, nullptr, TOKENS, DIMX, DIMX);

  // RoPE
  rope_f16<<<(TOKENS * NHEADS * 32) / 256, 256, 0, stream>>>(qf, kf);

  // sliding-window flash attention
  attention_wmma<<<dim3(SEQ / 128, NHEADS, BATCH), 256, 0, stream>>>(qf, kf, vf, ao);

  // out-proj + residual -> h (fp32)
  gemm_bf16_wmma<3, false><<<gproj, 256, 0, stream>>>(ao, wob, nullptr, hbuf, x, TOKENS, DIMX, DIMX);

  // LN2
  layernorm_bf16<<<TOKENS, 256, 0, stream>>>(hbuf, ln2g, ln2b, hn);

  // fused SwiGLU up-proj: silu(h@w1) * (h@w3) -> bf16
  gemm_bf16_wmma<0, true><<<dim3(HIDDEN / 64, TOKENS / 128), 256, 0, stream>>>(
      hn, w1b, w3b, gact, nullptr, TOKENS, HIDDEN, DIMX);

  // down-proj + residual -> fp32 output
  gemm_bf16_wmma<3, false><<<gproj, 256, 0, stream>>>(
      gact, w2b, nullptr, (float*)d_out, hbuf, TOKENS, DIMX, HIDDEN);
}
